// RNN_23390391894419
// MI455X (gfx1250) — matrix operand order repaired and checked by reading, not compiled, not run on hardware
//
#include <hip/hip_runtime.h>
#include <hip/hip_bf16.h>

// ---------------------------------------------------------------------------
// Model constants (match reference)
// ---------------------------------------------------------------------------
#define HIDDEN   512
#define LAYERS   2
#define MAXROLL  12
#define ANCHOR   64
#define BATCH    1024
#define GATES    2048            // 4*HIDDEN
#define KCAT     1024            // HIDDEN (x) + HIDDEN (h)
#define OUTCOLS  348             // 12 * (8+6+4+11)
#define CYCCOLS  29

typedef unsigned short u16;
typedef unsigned int   u32;

// WMMA vector types (CDNA5 / gfx1250, wave32)
typedef __attribute__((ext_vector_type(16))) __bf16 v16bf;
typedef __attribute__((ext_vector_type(8)))  float  v8f;

// ---------------------------------------------------------------------------
// Optional CDNA5 async global->LDS path (guarded; falls back to sync loads)
// Builtin signature (from diagnostics): void(v4i AS1*, v4i AS3*, imm offset, imm cpol)
// ---------------------------------------------------------------------------
#if defined(__HIP_DEVICE_COMPILE__) && __has_builtin(__builtin_amdgcn_global_load_async_to_lds_b128)
#define USE_ASYNC_LDS 1
typedef int vi4 __attribute__((vector_size(16)));
typedef __attribute__((address_space(1))) vi4* gp4;
typedef __attribute__((address_space(3))) vi4* lp4;
__device__ __forceinline__ void async_ld16(const void* g, void* l) {
    // generic LDS pointers on amdgcn carry the wave-relative LDS offset in the
    // low 32 bits; integer round-trip builds the AS(3)/AS(1) pointers.
    __builtin_amdgcn_global_load_async_to_lds_b128(
        (gp4)(uintptr_t)g, (lp4)(uintptr_t)l, 0, 0);
}
#endif

// ---------------------------------------------------------------------------
// Helpers
// ---------------------------------------------------------------------------
__device__ __forceinline__ u16 f32_to_bf16(float f) {
    union { float f; u32 u; } x; x.f = f;
    u32 u = x.u;
    u += 0x7FFFu + ((u >> 16) & 1u);   // round to nearest even
    return (u16)(u >> 16);
}
__device__ __forceinline__ float sigm(float x) { return 1.f / (1.f + expf(-x)); }

// ---------------------------------------------------------------------------
// Weight repack: Wcat_bf16[l][n][k] = bf16( k<512 ? Wih[l][n][k] : Whh[l][n][k-512] )
// ---------------------------------------------------------------------------
__global__ void k_wcat(const float* __restrict__ Wih, const float* __restrict__ Whh,
                       u16* __restrict__ Wcat) {
    size_t idx = (size_t)blockIdx.x * blockDim.x + threadIdx.x;   // < 2*2048*1024
    size_t l   = idx / ((size_t)GATES * KCAT);
    size_t rem = idx % ((size_t)GATES * KCAT);
    size_t n   = rem / KCAT;
    size_t k   = rem % KCAT;
    float v = (k < HIDDEN) ? Wih[(l * GATES + n) * HIDDEN + k]
                           : Whh[(l * GATES + n) * HIDDEN + (k - HIDDEN)];
    Wcat[idx] = f32_to_bf16(v);
}

__global__ void k_zero32(u32* __restrict__ p, size_t n) {
    size_t i = (size_t)blockIdx.x * blockDim.x + threadIdx.x;
    if (i < n) p[i] = 0u;
}

__global__ void k_x0cvt(const float* __restrict__ x0, u16* __restrict__ xcat0) {
    size_t i = (size_t)blockIdx.x * blockDim.x + threadIdx.x;  // < 1024*512
    size_t b = i >> 9, h = i & 511;
    xcat0[b * KCAT + h] = f32_to_bf16(x0[b * HIDDEN + h]);
}

// pp[0][b][a] = bprev[a]   (projection of the all-zero anchor state)
__global__ void k_pp0(const float* __restrict__ bprev, float* __restrict__ pp0) {
    size_t i = (size_t)blockIdx.x * blockDim.x + threadIdx.x;  // < 1024*64
    pp0[i] = bprev[i & 63];
}

// ---------------------------------------------------------------------------
// bf16 WMMA GEMM:  C[1024,2048] (f32) = A[1024,1024](bf16) @ W[2048,1024](bf16)^T
//   C[m,n] = sum_k A[m,k] * W[n,k]
// Block tile 128x128, 8 waves (2 M x 4 N), each wave 4x2 fragments of 16x16.
// K staged through double-buffered LDS in chunks of 32; stage s+1 is DMA'd
// (GLOBAL_LOAD_ASYNC_TO_LDS_B128, ASYNCcnt) under stage s's WMMAs.
// ---------------------------------------------------------------------------
#define TM 128
#define TN 128
#define TK 32
#define NSTG (KCAT / TK)

__global__ __launch_bounds__(256)
void k_gemm_bf16(const u16* __restrict__ A, const u16* __restrict__ W,
                 float* __restrict__ C) {
    __shared__ __align__(16) u16 sA[2][TM * TK];
    __shared__ __align__(16) u16 sB[2][TN * TK];

    const int tid  = threadIdx.x;
    const int wave = tid >> 5;       // 0..7
    const int lane = tid & 31;
    const int wm   = wave >> 2;      // 0..1  (64 rows each)
    const int wn   = wave & 3;       // 0..3  (32 cols each)
    const int bm   = blockIdx.y * TM;
    const int bn   = blockIdx.x * TN;
    const int lm   = lane & 15;      // row / col index within fragment
    const int lh   = lane >> 4;      // half-wave select

    // cooperative-load slot mapping: 512 x 16B chunks per tile, 2 per thread
    const int r0 = tid >> 2,         c0 = tid & 3;
    const int r1 = (tid + 256) >> 2, c1 = (tid + 256) & 3;

    v8f acc[4][2];
    #pragma unroll
    for (int fm = 0; fm < 4; ++fm)
        #pragma unroll
        for (int fn = 0; fn < 2; ++fn) {
            v8f z = {0.f,0.f,0.f,0.f,0.f,0.f,0.f,0.f};
            acc[fm][fn] = z;
        }

    union Frag { uint4 u[2]; v16bf v; };

    auto stage_issue = [&](int buf, int k0) {
#ifdef USE_ASYNC_LDS
        async_ld16(A + (size_t)(bm + r0) * KCAT + k0 + c0 * 8, &sA[buf][r0 * TK + c0 * 8]);
        async_ld16(A + (size_t)(bm + r1) * KCAT + k0 + c1 * 8, &sA[buf][r1 * TK + c1 * 8]);
        async_ld16(W + (size_t)(bn + r0) * KCAT + k0 + c0 * 8, &sB[buf][r0 * TK + c0 * 8]);
        async_ld16(W + (size_t)(bn + r1) * KCAT + k0 + c1 * 8, &sB[buf][r1 * TK + c1 * 8]);
#else
        uint4 a0 = *(const uint4*)(A + (size_t)(bm + r0) * KCAT + k0 + c0 * 8);
        uint4 a1 = *(const uint4*)(A + (size_t)(bm + r1) * KCAT + k0 + c1 * 8);
        uint4 b0 = *(const uint4*)(W + (size_t)(bn + r0) * KCAT + k0 + c0 * 8);
        uint4 b1 = *(const uint4*)(W + (size_t)(bn + r1) * KCAT + k0 + c1 * 8);
        *(uint4*)(&sA[buf][r0 * TK + c0 * 8]) = a0;
        *(uint4*)(&sA[buf][r1 * TK + c1 * 8]) = a1;
        *(uint4*)(&sB[buf][r0 * TK + c0 * 8]) = b0;
        *(uint4*)(&sB[buf][r1 * TK + c1 * 8]) = b1;
#endif
    };

    stage_issue(0, 0);   // prologue

    for (int s = 0; s < NSTG; ++s) {
        const int cur = s & 1;
        if (s + 1 < NSTG) {
            stage_issue(cur ^ 1, (s + 1) * TK);   // overlap with this stage's WMMAs
#ifdef USE_ASYNC_LDS
            asm volatile("s_wait_asynccnt 0x4" ::: "memory");  // stage-s DMAs done
#endif
        } else {
#ifdef USE_ASYNC_LDS
            asm volatile("s_wait_asynccnt 0x0" ::: "memory");
#endif
        }
        __syncthreads();   // all waves' stage-s tile is in LDS

        // ---- A fragments: 16x32 bf16, lanes 0-15 hold K 0-7 & 16-23 ----
        v16bf afrag[4];
        #pragma unroll
        for (int fm = 0; fm < 4; ++fm) {
            const u16* base = &sA[cur][(wm * 64 + fm * 16 + lm) * TK];
            Frag f;
            f.u[0] = *(const uint4*)(base + lh * 8);        // K = lh*8 .. +8
            f.u[1] = *(const uint4*)(base + 16 + lh * 8);   // K = 16+lh*8 .. +8
            afrag[fm] = f.v;
        }
        // ---- B fragments: 32x16 bf16, lanes 0-15 hold K 0-7 & 16-23 (col = lane) ----
        v16bf bfrag[2];
        #pragma unroll
        for (int fn = 0; fn < 2; ++fn) {
            const u16* base = &sB[cur][(wn * 32 + fn * 16 + lm) * TK + lh * 8];
            Frag f;
            f.u[0] = *(const uint4*)(base);
            f.u[1] = *(const uint4*)(base + 16);
            bfrag[fn] = f.v;
        }

        #pragma unroll
        for (int fm = 0; fm < 4; ++fm)
            #pragma unroll
            for (int fn = 0; fn < 2; ++fn)
                acc[fm][fn] = __builtin_amdgcn_wmma_f32_16x16x32_bf16(
                    false, afrag[fm], false, bfrag[fn],
                    (short)0, acc[fm][fn], false, false);

        __syncthreads();   // reads done before buf[cur] is overwritten at s+2
    }

    // ---- store: VGPR r holds row (lh*8 + r), col = lm ----
    #pragma unroll
    for (int fm = 0; fm < 4; ++fm) {
        int m0 = bm + wm * 64 + fm * 16 + lh * 8;
        #pragma unroll
        for (int fn = 0; fn < 2; ++fn) {
            int n = bn + wn * 32 + fn * 16 + lm;
            #pragma unroll
            for (int r = 0; r < 8; ++r)
                C[(size_t)(m0 + r) * GATES + n] = acc[fm][fn][r];
        }
    }
}

// ---------------------------------------------------------------------------
// LSTM gate fuse: torch gate order (i,f,g,o) over N=2048 = 4*512
// ---------------------------------------------------------------------------
__global__ void k_lstm_gate(const float* __restrict__ G,
                            const float* __restrict__ bih, const float* __restrict__ bhh,
                            float* __restrict__ c,
                            u16* __restrict__ h16a,   // stride KCAT, may be null
                            u16* __restrict__ h16b,   // stride KCAT, may be null
                            float* __restrict__ h32)  // stride HIDDEN, may be null
{
    size_t idx = (size_t)blockIdx.x * blockDim.x + threadIdx.x;  // < 1024*512
    size_t b = idx >> 9, j = idx & 511;
    const float* g = G + b * GATES;
    float gi = g[j]            + bih[j]            + bhh[j];
    float gf = g[512 + j]      + bih[512 + j]      + bhh[512 + j];
    float gg = g[1024 + j]     + bih[1024 + j]     + bhh[1024 + j];
    float go = g[1536 + j]     + bih[1536 + j]     + bhh[1536 + j];
    float cv = c[idx];
    cv = sigm(gf) * cv + sigm(gi) * tanhf(gg);
    float hv = sigm(go) * tanhf(cv);
    c[idx] = cv;
    u16 hb = f32_to_bf16(hv);
    if (h16a) h16a[b * KCAT + j] = hb;
    if (h16b) h16b[b * KCAT + j] = hb;
    if (h32)  h32[b * HIDDEN + j] = hv;
}

// ---------------------------------------------------------------------------
// Classifier head: logits = out @ Wc^T + bc ; softmax ; write into d_out slice
// ---------------------------------------------------------------------------
__global__ void k_head(const float* __restrict__ out, const float* __restrict__ Wc,
                       const float* __restrict__ bc, float* __restrict__ dout,
                       int S, int col) {
    int b = blockIdx.x * blockDim.x + threadIdx.x;
    if (b >= BATCH) return;
    const float* x = out + (size_t)b * HIDDEN;
    float logit[8];
    float mx = -3.4e38f;
    for (int s = 0; s < S; ++s) {
        const float* w = Wc + (size_t)s * HIDDEN;
        float acc = bc[s];
        for (int k = 0; k < HIDDEN; ++k) acc += x[k] * w[k];
        logit[s] = acc;
        mx = fmaxf(mx, acc);
    }
    float den = 0.f;
    for (int s = 0; s < S; ++s) { logit[s] = expf(logit[s] - mx); den += logit[s]; }
    float inv = 1.f / den;
    for (int s = 0; s < S; ++s) dout[(size_t)b * OUTCOLS + col + s] = logit[s] * inv;
}

// ---------------------------------------------------------------------------
// 64-wide projection: dst[b][a] = x[b,:] . Wp[a,:] + bp[a]
// ---------------------------------------------------------------------------
__global__ void k_proj64(const float* __restrict__ x, const float* __restrict__ Wp,
                         const float* __restrict__ bp, float* __restrict__ dst) {
    size_t idx = (size_t)blockIdx.x * blockDim.x + threadIdx.x;  // < 1024*64
    size_t b = idx >> 6, a = idx & 63;
    const float* xr = x + b * HIDDEN;
    const float* wr = Wp + a * HIDDEN;
    float acc = bp[a];
    for (int k = 0; k < HIDDEN; ++k) acc += xr[k] * wr[k];
    dst[idx] = acc;
}

// ---------------------------------------------------------------------------
// Anchor attention scores: ao[b][j] = j<cyc ? sigmoid(v . tanh(pp[j][b]+pc[b])) : 0
// ---------------------------------------------------------------------------
__global__ void k_anchor(const float* __restrict__ pp, const float* __restrict__ pc,
                         const float* __restrict__ v, int cyc,
                         float* __restrict__ ao, float* __restrict__ dout, int col) {
    size_t idx = (size_t)blockIdx.x * blockDim.x + threadIdx.x;  // < 1024*11
    size_t b = idx / (MAXROLL - 1), j = idx % (MAXROLL - 1);
    float s = 0.f;
    if ((int)j < cyc) {
        const float* p = pp + ((size_t)j * BATCH + b) * ANCHOR;
        const float* q = pc + b * ANCHOR;
        float acc = 0.f;
        for (int a = 0; a < ANCHOR; ++a) acc += v[a] * tanhf(p[a] + q[a]);
        s = sigm(acc);
    }
    ao[b * (MAXROLL - 1) + j] = s;
    dout[b * OUTCOLS + col + j] = s;
}

// ---------------------------------------------------------------------------
// Anchor embedding: inp[b][h] = ao[b,:] . Wanc[h,:] + banc[h]  -> bf16 into xcat0
// ---------------------------------------------------------------------------
__global__ void k_embed(const float* __restrict__ ao, const float* __restrict__ Wanc,
                        const float* __restrict__ banc, u16* __restrict__ xcat0) {
    size_t idx = (size_t)blockIdx.x * blockDim.x + threadIdx.x;  // < 1024*512
    size_t b = idx >> 9, h = idx & 511;
    const float* ar = ao + b * (MAXROLL - 1);
    const float* wr = Wanc + h * (MAXROLL - 1);
    float acc = banc[h];
    for (int j = 0; j < MAXROLL - 1; ++j) acc += ar[j] * wr[j];
    xcat0[b * KCAT + h] = f32_to_bf16(acc);
}

// ---------------------------------------------------------------------------
// Host orchestration
// ---------------------------------------------------------------------------
extern "C" void kernel_launch(void* const* d_in, const int* in_sizes, int n_in,
                              void* d_out, int out_size, void* d_ws, size_t ws_size,
                              hipStream_t stream) {
    const float* x0    = (const float*)d_in[0];
    const float* Wih   = (const float*)d_in[1];
    const float* Whh   = (const float*)d_in[2];
    const float* bih   = (const float*)d_in[3];
    const float* bhh   = (const float*)d_in[4];
    const float* Wc[3] = {(const float*)d_in[5], (const float*)d_in[7], (const float*)d_in[9]};
    const float* bc[3] = {(const float*)d_in[6], (const float*)d_in[8], (const float*)d_in[10]};
    const float* Wprev = (const float*)d_in[11];
    const float* bprev = (const float*)d_in[12];
    const float* Wcurr = (const float*)d_in[13];
    const float* bcurr = (const float*)d_in[14];
    const float* Wanc  = (const float*)d_in[15];
    const float* banc  = (const float*)d_in[16];
    const float* vvec  = (const float*)d_in[17];
    float* outp = (float*)d_out;

    // ---- workspace carve-up ----
    size_t off = 0;
    char* base = (char*)d_ws;
    auto carve = [&](size_t bytes) -> void* {
        void* p = base + off;
        off += (bytes + 255) & ~(size_t)255;
        return p;
    };
    u16*   wcat  = (u16*)  carve((size_t)LAYERS * GATES * KCAT * sizeof(u16)); // 8 MiB
    u16*   xcat0 = (u16*)  carve((size_t)BATCH * KCAT * sizeof(u16));          // 2 MiB
    u16*   xcat1 = (u16*)  carve((size_t)BATCH * KCAT * sizeof(u16));          // 2 MiB
    float* G     = (float*)carve((size_t)BATCH * GATES * sizeof(float));       // 8 MiB
    float* cst   = (float*)carve((size_t)LAYERS * BATCH * HIDDEN * sizeof(float)); // 4 MiB
    float* outb  = (float*)carve((size_t)BATCH * HIDDEN * sizeof(float));      // 2 MiB
    float* pp    = (float*)carve((size_t)MAXROLL * BATCH * ANCHOR * sizeof(float)); // 3 MiB
    float* pc    = (float*)carve((size_t)BATCH * ANCHOR * sizeof(float));
    float* ao    = (float*)carve((size_t)BATCH * (MAXROLL - 1) * sizeof(float));

    // ---- init (every call; deterministic, no caching) ----
    {
        size_t nwc = (size_t)LAYERS * GATES * KCAT;
        k_wcat<<<(unsigned)(nwc / 256), 256, 0, stream>>>(Wih, Whh, wcat);

        size_t nx = (size_t)BATCH * KCAT / 2;  // u32 count per xcat buffer
        k_zero32<<<(unsigned)((nx + 255) / 256), 256, 0, stream>>>((u32*)xcat0, nx);
        k_zero32<<<(unsigned)((nx + 255) / 256), 256, 0, stream>>>((u32*)xcat1, nx);

        size_t nc = (size_t)LAYERS * BATCH * HIDDEN;
        k_zero32<<<(unsigned)((nc + 255) / 256), 256, 0, stream>>>((u32*)cst, nc);

        size_t na = (size_t)BATCH * (MAXROLL - 1);
        k_zero32<<<(unsigned)((na + 255) / 256), 256, 0, stream>>>((u32*)ao, na);

        k_x0cvt<<<(BATCH * HIDDEN) / 256, 256, 0, stream>>>(x0, xcat0);
        k_pp0<<<(BATCH * ANCHOR) / 256, 256, 0, stream>>>(bprev, pp);
    }

    const int Ssz[3]  = {8, 6, 4};
    const int colo[3] = {0, 8, 14};
    const dim3 gemm_grid(GATES / TN, BATCH / TM);  // 16 x 8

    for (int i = 0; i < MAXROLL * 4; ++i) {
        int st = i & 3, cyc = i >> 2;

        // ---- layer 0: gates = [x|h0] @ Wcat0^T ----
        k_gemm_bf16<<<gemm_grid, 256, 0, stream>>>(xcat0, wcat, G);
        k_lstm_gate<<<(BATCH * HIDDEN) / 256, 256, 0, stream>>>(
            G, bih, bhh, cst,
            xcat0 + HIDDEN,   // h0 -> own recurrent slot
            xcat1,            // h0 -> layer1 input slot
            nullptr);

        // ---- layer 1: gates = [h0|h1] @ Wcat1^T ----
        k_gemm_bf16<<<gemm_grid, 256, 0, stream>>>(xcat1, wcat + (size_t)GATES * KCAT, G);
        k_lstm_gate<<<(BATCH * HIDDEN) / 256, 256, 0, stream>>>(
            G, bih + GATES, bhh + GATES, cst + (size_t)BATCH * HIDDEN,
            xcat1 + HIDDEN,   // h1 -> own recurrent slot
            nullptr,
            outb);            // h1 f32 for heads / attention

        if (st < 3) {
            k_head<<<BATCH / 256, 256, 0, stream>>>(
                outb, Wc[st], bc[st], outp, Ssz[st], cyc * CYCCOLS + colo[st]);
        } else {
            k_proj64<<<(BATCH * ANCHOR) / 256, 256, 0, stream>>>(outb, Wcurr, bcurr, pc);
            if (cyc + 1 < MAXROLL)  // projection of current anchor, for future cycles
                k_proj64<<<(BATCH * ANCHOR) / 256, 256, 0, stream>>>(
                    outb, Wprev, bprev, pp + (size_t)(cyc + 1) * BATCH * ANCHOR);
            k_anchor<<<(BATCH * (MAXROLL - 1)) / 256, 256, 0, stream>>>(
                pp, pc, vvec, cyc, ao, outp, cyc * CYCCOLS + 18);
            k_embed<<<(BATCH * HIDDEN) / 256, 256, 0, stream>>>(ao, Wanc, banc, xcat0);
        }
    }
}
